// NEUMF_Sample_40699110097047
// MI455X (gfx1250) — compile-verified, matrix operand order checked by reading, not run on hardware
//
#include <hip/hip_runtime.h>
#include <stdint.h>

// MI455X / gfx1250, wave32. bf16 WMMA 16x16x32, f32 accumulate.
//
// out[e] = sigmoid( relu([z[i]|z[j]]) @ W2 @ w3[0:128] + (z[i]*z[j]) @ w3[128:256] )
//
// d_in: [0]=X(1), [1]=train_edges(1M i32), [2]=train_false_edges(1M i32),
//       [3]=z (100000x128 f32), [4]=weight_two (256x128 f32), [5]=weight_three (256 f32)
// d_ws layout: [0,25.6MB) z in bf16; [25.6MB, +64KB) W2 packed bf16 (B-operand layout)

typedef __attribute__((ext_vector_type(16))) __bf16 v16bf;
typedef __attribute__((ext_vector_type(8)))  float  v8f;

#define E_TOTAL   1000000
#define E_TRUE    500000
#define NODE_D    128
#define ZB_ELEMS  (100000 * 128)
#define WPK_OFF   25600000  // byte offset of packed W2 in d_ws (64-byte aligned)

static __device__ __forceinline__ uint16_t f32_to_bf16_rne(float f) {
  unsigned u = __float_as_uint(f);
  unsigned r = ((u >> 16) & 1u) + 0x7FFFu;   // round-to-nearest-even
  return (uint16_t)((u + r) >> 16);
}

// --- z (f32) -> bf16 table, coalesced ---
__global__ void cvt_z_bf16(const float* __restrict__ z, uint16_t* __restrict__ zb) {
  int base = blockIdx.x * 2048;
  #pragma unroll
  for (int s = 0; s < 8; ++s) {
    int i = base + s * 256 + threadIdx.x;
    zb[i] = f32_to_bf16_rne(z[i]);
  }
}

// --- W2 (256x128 f32 row-major) -> bf16 packed for WMMA B-operand ds_load_b128 ---
// For K-step kk (32 K each): lane(nn,hi) needs column N, K-chunks [8*hi,+8) and
// [16+8*hi,+8). Chunk index c = (k&31)>>3, so pack element (k,n) at uint4 index
// (kk*4 + c)*128 + n, half h = k&7.
__global__ void pack_w2_bf16(const float* __restrict__ w2, uint16_t* __restrict__ wpk) {
  int t = blockIdx.x * 256 + threadIdx.x;   // 16 blocks * 256 = 4096 threads
  #pragma unroll
  for (int s = 0; s < 8; ++s) {
    int idx = s * 4096 + t;                 // 32768 elements total
    int k = idx >> 7, n = idx & 127;
    int kk = k >> 5, c = (k >> 3) & 3, h = k & 7;
    wpk[(size_t)(((kk * 4 + c) * 128 + n) * 8 + h)] = f32_to_bf16_rne(w2[idx]);
  }
}

union ABits { uint4 q[2]; unsigned u[8]; v16bf v; };

__global__ __launch_bounds__(256, 1) void neumf_fused_wmma(
    const int* __restrict__ te, const int* __restrict__ tf,
    const uint16_t* __restrict__ zb, const float* __restrict__ zf,
    const uint16_t* __restrict__ wpk, const float* __restrict__ w3,
    float* __restrict__ out)
{
  extern __shared__ unsigned char smem[];   // 64 KB: packed W2 in bf16

  { // cooperative stage: 4096 uint4 over 256 threads
    const uint4* src = (const uint4*)wpk;
    uint4* dst = (uint4*)smem;
    int t = threadIdx.x;
    #pragma unroll
    for (int r = 0; r < 16; ++r) dst[t + 256 * r] = src[t + 256 * r];
  }
  __syncthreads();

  const int lane = threadIdx.x & 31;
  const int wave = threadIdx.x >> 5;
  const int wt   = blockIdx.x * 8 + wave;   // wave-tile id: 32 edges each
  if (wt >= E_TOTAL / 32) return;           // uniform per wave; tail waves exit

  const int nn = lane & 15;
  const int hi = lane >> 4;
  const int e0 = wt * 32;

  // Edge rows for the two 16-edge M-tiles. 500000 is tile-aligned -> uniform select.
  const int eA = e0 + nn, eB = e0 + 16 + nn;
  int2 prA = (eA < E_TRUE) ? ((const int2*)te)[eA] : ((const int2*)tf)[eA - E_TRUE];
  int2 prB = (eB < E_TRUE) ? ((const int2*)te)[eB] : ((const int2*)tf)[eB - E_TRUE];

  const uint16_t* ziA = zb + (size_t)prA.x * NODE_D;
  const uint16_t* zjA = zb + (size_t)prA.y * NODE_D;
  const uint16_t* ziB = zb + (size_t)prB.x * NODE_D;
  const uint16_t* zjB = zb + (size_t)prB.y * NODE_D;

  float wt3[8];
  #pragma unroll
  for (int n = 0; n < 8; ++n) wt3[n] = w3[n * 16 + nn];  // w3_top[N], N = 16n+nn

  const v8f vzero = {0.f, 0.f, 0.f, 0.f, 0.f, 0.f, 0.f, 0.f};
  v8f acc0[8], acc1[8];
  #pragma unroll
  for (int n = 0; n < 8; ++n) { acc0[n] = vzero; acc1[n] = vzero; }

  // K loop: K-step kk covers K = 32*kk..32*kk+31; K<128 comes from zi, else zj.
  for (int kk = 0; kk < 8; ++kk) {
    const int koff = (kk & 3) * 32 + hi * 8;   // halves within the 128-wide row
    const uint16_t* rA = (kk < 4) ? ziA : zjA;
    const uint16_t* rB = (kk < 4) ? ziB : zjB;

    ABits a0, a1;
    a0.q[0] = *(const uint4*)(rA + koff);
    a0.q[1] = *(const uint4*)(rA + koff + 16);
    a1.q[0] = *(const uint4*)(rB + koff);
    a1.q[1] = *(const uint4*)(rB + koff + 16);

    // relu on packed bf16: zero each 16-bit half whose sign bit is set
    #pragma unroll
    for (int wd = 0; wd < 8; ++wd) {
      unsigned sa = a0.u[wd] & 0x80008000u;
      a0.u[wd] &= ~((sa - (sa >> 15)) | sa);
      unsigned sb = a1.u[wd] & 0x80008000u;
      a1.u[wd] &= ~((sb - (sb >> 15)) | sb);
    }

    const uint4* bl = (const uint4*)smem;
    #pragma unroll
    for (int n = 0; n < 8; ++n) {
      ABits b;
      b.q[0] = bl[(kk * 4 + hi) * 128 + n * 16 + nn];
      b.q[1] = bl[(kk * 4 + 2 + hi) * 128 + n * 16 + nn];
      acc0[n] = __builtin_amdgcn_wmma_f32_16x16x32_bf16(
          false, a0.v, false, b.v, (short)0, acc0[n], false, false);
      acc1[n] = __builtin_amdgcn_wmma_f32_16x16x32_bf16(
          false, a1.v, false, b.v, (short)0, acc1[n], false, false);
    }
  }

  // Fold h @ w3_top: lane(nn,hi) holds h[M=r+8*hi][N=16n+nn] in acc[n][r].
  float s0[8], s1[8];
  #pragma unroll
  for (int r = 0; r < 8; ++r) { s0[r] = 0.f; s1[r] = 0.f; }
  #pragma unroll
  for (int n = 0; n < 8; ++n) {
    #pragma unroll
    for (int r = 0; r < 8; ++r) {
      s0[r] = fmaf(acc0[n][r], wt3[n], s0[r]);
      s1[r] = fmaf(acc1[n][r], wt3[n], s1[r]);
    }
  }
  // reduce over the 16 N-lanes of each half-wave
  #pragma unroll
  for (int m = 1; m <= 8; m <<= 1) {
    #pragma unroll
    for (int r = 0; r < 8; ++r) {
      s0[r] += __shfl_xor(s0[r], m, 32);
      s1[r] += __shfl_xor(s1[r], m, 32);
    }
  }

  // Elementwise part in exact f32: lane(nn,hi) covers row nn, k in [64*hi, 64*hi+64)
  const float* fiA = zf + (size_t)prA.x * NODE_D + hi * 64;
  const float* fjA = zf + (size_t)prA.y * NODE_D + hi * 64;
  const float* fiB = zf + (size_t)prB.x * NODE_D + hi * 64;
  const float* fjB = zf + (size_t)prB.y * NODE_D + hi * 64;
  const float* w3b = w3 + 128 + hi * 64;
  float ep0 = 0.f, ep1 = 0.f;
  #pragma unroll 4
  for (int q = 0; q < 16; ++q) {
    float4 w4 = *(const float4*)(w3b + q * 4);
    float4 aA = *(const float4*)(fiA + q * 4);
    float4 bA = *(const float4*)(fjA + q * 4);
    float4 aB = *(const float4*)(fiB + q * 4);
    float4 bB = *(const float4*)(fjB + q * 4);
    ep0 = fmaf(aA.x * bA.x, w4.x, ep0); ep0 = fmaf(aA.y * bA.y, w4.y, ep0);
    ep0 = fmaf(aA.z * bA.z, w4.z, ep0); ep0 = fmaf(aA.w * bA.w, w4.w, ep0);
    ep1 = fmaf(aB.x * bB.x, w4.x, ep1); ep1 = fmaf(aB.y * bB.y, w4.y, ep1);
    ep1 = fmaf(aB.z * bB.z, w4.z, ep1); ep1 = fmaf(aB.w * bB.w, w4.w, ep1);
  }
  ep0 += __shfl_xor(ep0, 16, 32);  // combine the two k-halves of the same row
  ep1 += __shfl_xor(ep1, 16, 32);

  // Writer lanes: hi == nn>>3 -> lane's register set contains row M == nn
  // (M = r + 8*hi with r = nn&7). Each writer emits both M-tiles (2 edges).
  const int rr = nn & 7;
  float v0 = s0[0], v1 = s1[0];
  #pragma unroll
  for (int r = 1; r < 8; ++r) {
    v0 = (rr == r) ? s0[r] : v0;
    v1 = (rr == r) ? s1[r] : v1;
  }
  if (hi == (nn >> 3)) {
    float l0 = v0 + ep0;
    float l1 = v1 + ep1;
    out[e0 + nn]      = 1.0f / (1.0f + __expf(-l0));
    out[e0 + 16 + nn] = 1.0f / (1.0f + __expf(-l1));
  }
}

extern "C" void kernel_launch(void* const* d_in, const int* in_sizes, int n_in,
                              void* d_out, int out_size, void* d_ws, size_t ws_size,
                              hipStream_t stream) {
  const int*   te = (const int*)d_in[1];
  const int*   tf = (const int*)d_in[2];
  const float* z  = (const float*)d_in[3];
  const float* w2 = (const float*)d_in[4];
  const float* w3 = (const float*)d_in[5];
  float* out = (float*)d_out;

  uint16_t* zb  = (uint16_t*)d_ws;                        // 25.6 MB bf16 z
  uint16_t* wpk = (uint16_t*)((char*)d_ws + WPK_OFF);     // 64 KB packed W2

  // allow 64 KB dynamic LDS (idempotent; safe under graph capture)
  (void)hipFuncSetAttribute((const void*)neumf_fused_wmma,
                            hipFuncAttributeMaxDynamicSharedMemorySize, 65536);

  cvt_z_bf16<<<ZB_ELEMS / 2048, 256, 0, stream>>>(z, zb);
  pack_w2_bf16<<<16, 256, 0, stream>>>(w2, wpk);

  const int nblocks = (E_TOTAL / 32 + 7) / 8;             // 3907
  neumf_fused_wmma<<<nblocks, 256, 65536, stream>>>(te, tf, zb, z, wpk, w3, out);
}
